// LOFTR_85770496901423
// MI455X (gfx1250) — matrix operand order repaired and checked by reading, not compiled, not run on hardware
//
#include <hip/hip_runtime.h>
#include <hip/hip_bf16.h>

// ---------------------------------------------------------------------------
// LoFTR encoder stack on gfx1250: all dense GEMMs via v_wmma_f32_16x16x32_f16
// (f16 in, f32 accumulate), double-buffered LDS, linear attention + LayerNorm
// in f32 VALU.
// ---------------------------------------------------------------------------

typedef __attribute__((ext_vector_type(4)))  float    floatx4;
typedef __attribute__((ext_vector_type(8)))  float    v8f;
typedef __attribute__((ext_vector_type(8)))  _Float16 halfx8;
typedef __attribute__((ext_vector_type(16))) _Float16 halfx16;

#define TM 128         // block tile rows
#define TNB 128        // block tile cols
#define TK 32          // K-step
#define SA 40          // padded LDS row stride in halves (80B, 16B aligned)

static constexpr int NB  = 4;            // batch
static constexpr int LSEQ = 4800;        // tokens per image
static constexpr int MROWS = NB * LSEQ;  // 19200
static constexpr int DMODEL = 256;

__device__ __forceinline__ halfx8 pack8(floatx4 u, floatx4 v) {
    halfx8 r = { (_Float16)u[0], (_Float16)u[1], (_Float16)u[2], (_Float16)u[3],
                 (_Float16)v[0], (_Float16)v[1], (_Float16)v[2], (_Float16)v[3] };
    return r;
}

// ---------------------------------------------------------------------------
// GEMM: Y[M,NN] = act( concat(X0[:,0:K0], X1[:,0:KTOT-K0]) @ W[NN,KTOT]^T )
// ACT: 0 = none, 1 = relu, 2 = elu(x)+1
// Block: 256 threads (8 waves), tile 128x128, double-buffered LDS,
// one barrier per K-step.  Wave tile 32x64 -> acc[2][4] (8 WMMAs / step).
// ---------------------------------------------------------------------------
template<int ACT, int K0, int KTOT, int NN>
__global__ __launch_bounds__(256) void gemm_wmma(
    const float* __restrict__ X0, const float* __restrict__ X1,
    const float* __restrict__ W, float* __restrict__ Y)
{
    constexpr int NKSTEP = KTOT / TK;
    __shared__ _Float16 As[2][TM  * SA];
    __shared__ _Float16 Bs[2][TNB * SA];

    const int t    = threadIdx.x;
    const int lane = t & 31;
    const int wid  = t >> 5;
    const int wm   = wid & 3;        // 32-row group (0..3)
    const int wn   = wid >> 2;       // 64-col group (0..1)
    const long mbase = (long)blockIdx.x * TM;
    const int  nbase = blockIdx.y * TNB;

    const v8f vzero = {0.f,0.f,0.f,0.f,0.f,0.f,0.f,0.f};
    v8f acc[2][4];
    #pragma unroll
    for (int ms = 0; ms < 2; ++ms)
        #pragma unroll
        for (int ns = 0; ns < 4; ++ns) acc[ms][ns] = vzero;

    // Staging: both tiles are 128 rows x 32 cols; 2 threads per row,
    // each thread stages 16 contiguous floats -> 16 halves.
    const int srow  = t >> 1;          // 0..127
    const int shalf = (t & 1) * 16;    // 0 / 16
    const int mlo   = lane & 15;
    const int ksel  = lane >> 4;       // 0 / 1

    // ---- prologue: stage K-step 0 into buffer 0 ----
    {
        const float* ga = X0 + (mbase + srow) * (long)K0 + shalf;
        floatx4 a0 = ((const floatx4*)ga)[0];
        floatx4 a1 = ((const floatx4*)ga)[1];
        floatx4 a2 = ((const floatx4*)ga)[2];
        floatx4 a3 = ((const floatx4*)ga)[3];
        const float* gb = W + (long)(nbase + srow) * KTOT + shalf;
        floatx4 b0 = ((const floatx4*)gb)[0];
        floatx4 b1 = ((const floatx4*)gb)[1];
        floatx4 b2 = ((const floatx4*)gb)[2];
        floatx4 b3 = ((const floatx4*)gb)[3];
        *(halfx8*)&As[0][srow * SA + shalf]     = pack8(a0, a1);
        *(halfx8*)&As[0][srow * SA + shalf + 8] = pack8(a2, a3);
        *(halfx8*)&Bs[0][srow * SA + shalf]     = pack8(b0, b1);
        *(halfx8*)&Bs[0][srow * SA + shalf + 8] = pack8(b2, b3);
    }
    __syncthreads();

    int p = 0;
    for (int ks = 0; ks < NKSTEP; ++ks) {
        // ---- issue next tile's global loads early (hide behind WMMAs) ----
        floatx4 ra0, ra1, ra2, ra3, rb0, rb1, rb2, rb3;
        const bool havenext = (ks + 1 < NKSTEP);
        if (havenext) {
            const int kc = (ks + 1) * TK;
            const float* gan;
            if (kc < K0) gan = X0 + (mbase + srow) * (long)K0 + kc + shalf;
            else         gan = X1 + (mbase + srow) * (long)(KTOT - K0) + (kc - K0) + shalf;
            ra0 = ((const floatx4*)gan)[0];
            ra1 = ((const floatx4*)gan)[1];
            ra2 = ((const floatx4*)gan)[2];
            ra3 = ((const floatx4*)gan)[3];
            const float* gbn = W + (long)(nbase + srow) * KTOT + kc + shalf;
            rb0 = ((const floatx4*)gbn)[0];
            rb1 = ((const floatx4*)gbn)[1];
            rb2 = ((const floatx4*)gbn)[2];
            rb3 = ((const floatx4*)gbn)[3];
        }

        // ---- compute from buffer p ----
        halfx16 afrag[2];
        // A fragment: lane holds row (lane&15); k = ksel*8 + j, then +16.
        #pragma unroll
        for (int ms = 0; ms < 2; ++ms) {
            const _Float16* ap = &As[p][(wm*32 + ms*16 + mlo) * SA + ksel*8];
            halfx8 lo = *(const halfx8*)ap;
            halfx8 hi = *(const halfx8*)(ap + 16);
            afrag[ms] = __builtin_shufflevector(lo, hi,
                0,1,2,3,4,5,6,7,8,9,10,11,12,13,14,15);
        }
        // B fragment: lane holds column n=(lane&15); k = ksel*8 + j, then +16.
        #pragma unroll
        for (int ns = 0; ns < 4; ++ns) {
            const _Float16* bp = &Bs[p][(wn*64 + ns*16 + mlo) * SA + ksel*8];
            halfx8 lo = *(const halfx8*)bp;
            halfx8 hi = *(const halfx8*)(bp + 16);
            halfx16 bfrag = __builtin_shufflevector(lo, hi,
                0,1,2,3,4,5,6,7,8,9,10,11,12,13,14,15);
            #pragma unroll
            for (int ms = 0; ms < 2; ++ms)
                acc[ms][ns] = __builtin_amdgcn_wmma_f32_16x16x32_f16(
                    false, afrag[ms], false, bfrag,
                    (short)0, acc[ms][ns], false, false);
        }

        // ---- convert + store next tile into the other buffer ----
        if (havenext) {
            const int q = p ^ 1;
            *(halfx8*)&As[q][srow * SA + shalf]     = pack8(ra0, ra1);
            *(halfx8*)&As[q][srow * SA + shalf + 8] = pack8(ra2, ra3);
            *(halfx8*)&Bs[q][srow * SA + shalf]     = pack8(rb0, rb1);
            *(halfx8*)&Bs[q][srow * SA + shalf + 8] = pack8(rb2, rb3);
        }
        __syncthreads();
        p ^= 1;
    }

    // D layout: VGPR i -> m = (lane>>4)*8 + i, n = lane&15
    const int mtop = ksel * 8;
    #pragma unroll
    for (int ms = 0; ms < 2; ++ms) {
        #pragma unroll
        for (int ns = 0; ns < 4; ++ns) {
            long mrow = mbase + wm*32 + ms*16 + mtop;
            int  ncol = nbase + wn*64 + ns*16 + mlo;
            float* yp = Y + mrow * NN + ncol;
            #pragma unroll
            for (int i = 0; i < 8; ++i) {
                float v = acc[ms][ns][i];
                if (ACT == 1) v = fmaxf(v, 0.f);
                if (ACT == 2) v = (v > 0.f) ? (v + 1.f) : __expf(v);
                yp[(long)i * NN] = v;
            }
        }
    }
}

// ---------------------------------------------------------------------------
// KV[n,h] = sum_s K[n,s,h,:] outer (V[n,s,h,:]/L) ; Ksum[n,h] = sum_s K[n,s,h,:]
// One block per (n,h). Thread t: d = t>>3, e0 = (t&7)*4 (4 outputs each).
// ---------------------------------------------------------------------------
__global__ __launch_bounds__(256) void kv_kernel(
    const float* __restrict__ Kf, const float* __restrict__ V,
    float* __restrict__ KV, float* __restrict__ Ksum, int L, float inv_slen)
{
    __shared__ float Kc[64 * 36];
    __shared__ float Vc[64 * 36];
    const int n = blockIdx.x >> 3, h = blockIdx.x & 7;
    const int t = threadIdx.x;
    const int d = t >> 3, e0 = (t & 7) * 4;
    const int r = t >> 2, c0 = (t & 3) * 8;
    const long rowbase = (long)n * L;

    float acc0 = 0.f, acc1 = 0.f, acc2 = 0.f, acc3 = 0.f, ks = 0.f;

    for (int s0 = 0; s0 < L; s0 += 64) {
        const float* gk = Kf + (rowbase + s0 + r) * DMODEL + h*32 + c0;
        const float* gv = V  + (rowbase + s0 + r) * DMODEL + h*32 + c0;
        floatx4 k0 = ((const floatx4*)gk)[0], k1 = ((const floatx4*)gk)[1];
        floatx4 v0 = ((const floatx4*)gv)[0], v1 = ((const floatx4*)gv)[1];
        v0 *= inv_slen; v1 *= inv_slen;
        *(floatx4*)&Kc[r*36 + c0]     = k0;
        *(floatx4*)&Kc[r*36 + c0 + 4] = k1;
        *(floatx4*)&Vc[r*36 + c0]     = v0;
        *(floatx4*)&Vc[r*36 + c0 + 4] = v1;
        __syncthreads();
        #pragma unroll 8
        for (int s = 0; s < 64; ++s) {
            float kd = Kc[s*36 + d];
            ks += kd;
            const float* vr = &Vc[s*36 + e0];
            acc0 += kd * vr[0]; acc1 += kd * vr[1];
            acc2 += kd * vr[2]; acc3 += kd * vr[3];
        }
        __syncthreads();
    }
    long ob = ((long)(n*8 + h) * 32 + d) * 32 + e0;
    KV[ob] = acc0; KV[ob+1] = acc1; KV[ob+2] = acc2; KV[ob+3] = acc3;
    if (e0 == 0) Ksum[(n*8 + h)*32 + d] = ks;
}

// ---------------------------------------------------------------------------
// msg[n,l,h,:] = (Q[n,l,h,:] @ KV[n,h]) * s_len / (Q.Ksum + eps)
// grid = (L/32, N); thread t: token lt=t>>3, head h=t&7.
// ---------------------------------------------------------------------------
__global__ __launch_bounds__(256) void msg_kernel(
    const float* __restrict__ Q, const float* __restrict__ KV,
    const float* __restrict__ Ksum, float* __restrict__ Msg, int L, float slen)
{
    __shared__ float KVsh[8 * 1032];   // per-head stride 1032 -> bank de-conflict
    __shared__ float Ks[256];
    const int n = blockIdx.y;
    const int t = threadIdx.x;
    {
        int hh = t >> 5, idx = t & 31;
        const float* src = KV + ((long)(n*8 + hh) * 32 + idx) * 32;
        float* dst = &KVsh[hh * 1032 + idx * 32];
        #pragma unroll
        for (int j = 0; j < 8; ++j) ((floatx4*)dst)[j] = ((const floatx4*)src)[j];
        Ks[t] = Ksum[n * 256 + t];
    }
    __syncthreads();

    const int lt = t >> 3, h = t & 7;
    const long row = (long)n * L + (long)blockIdx.x * 32 + lt;
    const float* qp = Q + row * DMODEL + h * 32;
    float q[32];
    #pragma unroll
    for (int j = 0; j < 8; ++j) {
        floatx4 v = ((const floatx4*)qp)[j];
        q[j*4] = v[0]; q[j*4+1] = v[1]; q[j*4+2] = v[2]; q[j*4+3] = v[3];
    }
    float zden = 1e-6f;
    #pragma unroll
    for (int dd = 0; dd < 32; ++dd) zden += q[dd] * Ks[h*32 + dd];
    const float fac = slen / zden;

    float acc[32];
    #pragma unroll
    for (int e = 0; e < 32; ++e) acc[e] = 0.f;
    const float* kvh = &KVsh[h * 1032];
    #pragma unroll 4
    for (int dd = 0; dd < 32; ++dd) {
        float qd = q[dd];
        #pragma unroll
        for (int e = 0; e < 32; ++e) acc[e] += qd * kvh[dd*32 + e];
    }
    float* mp = Msg + row * DMODEL + h * 32;
    #pragma unroll
    for (int e = 0; e < 32; ++e) mp[e] = acc[e] * fac;
}

// ---------------------------------------------------------------------------
// LayerNorm over last dim (256). One wave per row (wave32, __shfl_xor reduce).
// Y = LN(Xin)*g + b  (+ resid if non-null).  May run in-place (Y == resid).
// ---------------------------------------------------------------------------
__global__ __launch_bounds__(256) void ln_kernel(
    const float* __restrict__ Xin, const float* __restrict__ g,
    const float* __restrict__ b, const float* resid, float* Y)
{
    const int t = threadIdx.x, lane = t & 31, w = t >> 5;
    const long row = (long)blockIdx.x * 8 + w;
    const float* xp = Xin + row * DMODEL;
    float v[8];
    float s = 0.f;
    #pragma unroll
    for (int j = 0; j < 8; ++j) { v[j] = xp[j*32 + lane]; s += v[j]; }
    #pragma unroll
    for (int off = 16; off > 0; off >>= 1) s += __shfl_xor(s, off, 32);
    const float m = s * (1.f / 256.f);
    float vs = 0.f;
    #pragma unroll
    for (int j = 0; j < 8; ++j) { float d = v[j] - m; vs += d * d; }
    #pragma unroll
    for (int off = 16; off > 0; off >>= 1) vs += __shfl_xor(vs, off, 32);
    const float rs = rsqrtf(vs * (1.f / 256.f) + 1e-5f);
    float* yp = Y + row * DMODEL;
    const float* rp = resid ? resid + row * DMODEL : nullptr;
    #pragma unroll
    for (int j = 0; j < 8; ++j) {
        int c = j*32 + lane;
        float o = (v[j] - m) * rs * g[c] + b[c];
        if (rp) o += rp[c];
        yp[c] = o;
    }
}

// ---------------------------------------------------------------------------
// Host-side orchestration
// ---------------------------------------------------------------------------
struct Ws {
    float *bQ, *bK, *bV, *bMsg, *bT, *bY1, *bKV, *bKs;
};

static void run_encoder(float* x, const float* src,
                        const float* wq, const float* wk, const float* wv,
                        const float* wm, const float* w1, const float* w2,
                        const float* g1, const float* b1,
                        const float* g2, const float* b2,
                        const Ws& W, hipStream_t stream)
{
    const dim3 blk(256);
    const dim3 g256(MROWS / TM, 256 / TNB);   // (150, 2)
    const dim3 g512(MROWS / TM, 512 / TNB);   // (150, 4)

    // Q = elu(x@Wq^T)+1 ; K = elu(src@Wk^T)+1 ; V = src@Wv^T
    gemm_wmma<2, 256, 256, 256><<<g256, blk, 0, stream>>>(x,   nullptr, wq, W.bQ);
    gemm_wmma<2, 256, 256, 256><<<g256, blk, 0, stream>>>(src, nullptr, wk, W.bK);
    gemm_wmma<0, 256, 256, 256><<<g256, blk, 0, stream>>>(src, nullptr, wv, W.bV);

    kv_kernel<<<NB * 8, blk, 0, stream>>>(W.bK, W.bV, W.bKV, W.bKs, LSEQ,
                                          1.0f / (float)LSEQ);
    msg_kernel<<<dim3(LSEQ / 32, NB), blk, 0, stream>>>(W.bQ, W.bKV, W.bKs, W.bMsg,
                                                        LSEQ, (float)LSEQ);

    // merge + LN1 (LN result reuses bMsg)
    gemm_wmma<0, 256, 256, 256><<<g256, blk, 0, stream>>>(W.bMsg, nullptr, wm, W.bT);
    ln_kernel<<<MROWS / 8, blk, 0, stream>>>(W.bT, g1, b1, nullptr, W.bMsg);

    // MLP: relu([x, ln1] @ W1^T) @ W2^T, LN2 + residual (in place on x)
    gemm_wmma<1, 256, 512, 512><<<g512, blk, 0, stream>>>(x, W.bMsg, w1, W.bY1);
    gemm_wmma<0, 512, 512, 256><<<g256, blk, 0, stream>>>(W.bY1, nullptr, w2, W.bT);
    ln_kernel<<<MROWS / 8, blk, 0, stream>>>(W.bT, g2, b2, x, x);
}

extern "C" void kernel_launch(void* const* d_in, const int* in_sizes, int n_in,
                              void* d_out, int out_size, void* d_ws, size_t ws_size,
                              hipStream_t stream)
{
    (void)in_sizes; (void)n_in; (void)out_size; (void)ws_size;
    const float* feat0 = (const float*)d_in[0];
    const float* feat1 = (const float*)d_in[1];
    const float* Wq = (const float*)d_in[2];
    const float* Wk = (const float*)d_in[3];
    const float* Wv = (const float*)d_in[4];
    const float* Wm = (const float*)d_in[5];
    const float* W1 = (const float*)d_in[6];
    const float* W2 = (const float*)d_in[7];
    const float* G1 = (const float*)d_in[8];
    const float* B1 = (const float*)d_in[9];
    const float* G2 = (const float*)d_in[10];
    const float* B2 = (const float*)d_in[11];

    const size_t feat_elems = (size_t)MROWS * DMODEL;
    float* out = (float*)d_out;
    float* f0 = out;
    float* f1 = out + feat_elems;
    hipMemcpyAsync(f0, feat0, feat_elems * sizeof(float), hipMemcpyDeviceToDevice, stream);
    hipMemcpyAsync(f1, feat1, feat_elems * sizeof(float), hipMemcpyDeviceToDevice, stream);

    // Workspace layout (f32).  bY1 (M x 512) overlaps bQ+bK, which are dead
    // by the time the MLP runs.  Total: 5 * M*256 + KV + Ksum ~= 98 MB.
    float* ws = (float*)d_ws;
    Ws W;
    W.bQ   = ws;
    W.bK   = W.bQ   + feat_elems;
    W.bV   = W.bK   + feat_elems;
    W.bMsg = W.bV   + feat_elems;
    W.bT   = W.bMsg + feat_elems;
    W.bY1  = W.bQ;                       // reuse Q+K region (M x 512)
    W.bKV  = W.bT   + feat_elems;
    W.bKs  = W.bKV  + (size_t)NB * 8 * 32 * 32;

    for (int i = 0; i < 8; ++i) {
        const float* wq = Wq + (size_t)i * 256 * 256;
        const float* wk = Wk + (size_t)i * 256 * 256;
        const float* wv = Wv + (size_t)i * 256 * 256;
        const float* wm = Wm + (size_t)i * 256 * 256;
        const float* w1 = W1 + (size_t)i * 512 * 512;
        const float* w2 = W2 + (size_t)i * 256 * 512;
        const float* g1 = G1 + (size_t)i * 256;
        const float* b1 = B1 + (size_t)i * 256;
        const float* g2 = G2 + (size_t)i * 256;
        const float* b2 = B2 + (size_t)i * 256;
        if ((i & 1) == 0) {   // self attention
            run_encoder(f0, f0, wq, wk, wv, wm, w1, w2, g1, b1, g2, b2, W, stream);
            run_encoder(f1, f1, wq, wk, wv, wm, w1, w2, g1, b1, g2, b2, W, stream);
        } else {              // cross attention (f1 sees updated f0)
            run_encoder(f0, f1, wq, wk, wv, wm, w1, w2, g1, b1, g2, b2, W, stream);
            run_encoder(f1, f0, wq, wk, wv, wm, w1, w2, g1, b1, g2, b2, W, stream);
        }
    }
}
